// TwoDGRUClassifier_1211180777485
// MI455X (gfx1250) — compile-verified, matrix operand order checked by reading, not run on hardware
//
#include <hip/hip_runtime.h>
#include <hip/hip_bf16.h>
#include <cmath>

// ---------------------------------------------------------------------------
// 2D GRU classifier for MI455X (gfx1250, wave32, WMMA).
//
// pre[b,i,j,f] = x[b,i,j]*v[f] + c[f]        (C==1 folds the embedding)
// per cell:  m = h_prev @ [U_zr | U_n]  via v_wmma_f32_16x16x32_f16
//            z = sig(pre_z + m_z); r = sig(pre_r + m_r)
//            n = tanh(pre_n + r*m_n); h = (1-z)*n + z*h_prev
//
// 55-diagonal wavefront inside ONE persistent kernel (112 blocks) with a
// device-wide release/acquire barrier between diagonals. Each wave computes
// a 32x32 register tile (2 M-frags x 2 N-frags x 3 gates = 12 WMMA accs),
// halving per-WMMA fragment-load traffic vs 1x1 tiling so the kernel is
// matrix-unit bound instead of LDS/L2 bound. U (1.5 MB f16) and h state
// (14.7 MB) stay L2-resident (192 MB L2).
// ---------------------------------------------------------------------------

#define B_        128
#define Hh        28
#define Ww        28
#define E_        64
#define HD        512
#define NC        10
#define THREE_HD  1536
#define LDA       520   // padded f16 stride for LDS A tile (row = 1040 B, 16B aligned)
#define NBLK      (Ww * 4)   // 28 cell slots x 4 N-slices = 112 persistent blocks
#define NDIAG     (Hh + Ww - 1)

typedef __attribute__((ext_vector_type(16))) _Float16 v16h;
typedef __attribute__((ext_vector_type(8)))  _Float16 v8h_t;
typedef __attribute__((ext_vector_type(4)))  _Float16 v4h_t;
typedef __attribute__((ext_vector_type(8)))  float    v8f;

// 16-bit A/B fragment gather per CDNA5 ISA 7.12.2:
// lane L: row = L&15, K in [khalf*8, +8) and [16+khalf*8, +8)  (khalf = L>>4)
// => two contiguous 16-byte chunks per fragment (b128 loads).
__device__ __forceinline__ v16h frag_from(const _Float16* __restrict__ base, int k0) {
  v8h_t lo = *(const v8h_t*)(base + k0);
  v8h_t hi = *(const v8h_t*)(base + k0 + 16);
  v16h f;
#pragma unroll
  for (int t = 0; t < 8; ++t) { f[t] = lo[t]; f[8 + t] = hi[t]; }
  return f;
}

// ---- prep: fold embedding + W_x into v[1536], c[1536] ----------------------
__global__ void fold_kernel(const float* __restrict__ W_emb, const float* __restrict__ b_emb,
                            const float* __restrict__ W_x,   const float* __restrict__ b_x,
                            float* __restrict__ vfold, float* __restrict__ cfold) {
  int f = blockIdx.x * blockDim.x + threadIdx.x;
  if (f >= THREE_HD) return;
  float v = 0.f, c = 0.f;
#pragma unroll 8
  for (int k = 0; k < E_; ++k) {
    float w = W_x[(size_t)k * THREE_HD + f];
    v += W_emb[k] * w;
    c += b_emb[k] * w;
  }
  vfold[f] = v;
  cfold[f] = c + b_x[f];
}

// ---- prep: transpose+convert [U_zr | U_n] -> Ut[n=1536][k=512] f16 ---------
__global__ void pack_u_kernel(const float* __restrict__ U_zr, const float* __restrict__ U_n,
                              _Float16* __restrict__ Ut) {
  int idx = blockIdx.x * blockDim.x + threadIdx.x;
  if (idx >= THREE_HD * HD) return;
  int n = idx >> 9;        // / 512
  int k = idx & 511;
  float s = (n < 2 * HD) ? U_zr[(size_t)k * (2 * HD) + n]
                         : U_n [(size_t)k * HD + (n - 2 * HD)];
  Ut[(size_t)n * HD + k] = (_Float16)s;
}

// ---- persistent wavefront kernel ------------------------------------------
__global__ __launch_bounds__(256) void gru_wavefront_kernel(
    const float* __restrict__ x,      // (B,28,28)
    const float* __restrict__ vfold,  // (1536)
    const float* __restrict__ cfold,  // (1536)
    const _Float16* __restrict__ Ut,  // (1536,512) f16, row = output col
    float* __restrict__ Hbuf0,        // (28,B,HD) parity 0
    float* __restrict__ Hbuf1,        // (28,B,HD) parity 1
    unsigned* __restrict__ bar)       // device-wide barrier counter (pre-zeroed)
{
  __shared__ _Float16 As[B_ * LDA];   // 130 KB of the 320 KB WGP LDS

  const int slice    = blockIdx.x & 3;       // N-slice: 128 h-columns
  const int cellslot = blockIdx.x >> 2;      // 0..27 -> column offset on diagonal
  const int f0  = slice * 128;
  const int tid = threadIdx.x;
  const int wave  = tid >> 5;
  const int lane  = tid & 31;
  const int l15   = lane & 15;
  const int khalf = lane >> 4;
  // 2x2 register tiling: wave owns M-group (w&3) = rows mg*32..+32,
  // iterates N-groups {2*(w>>2), 2*(w>>2)+1} = cols ng*32..+32 of the slice.
  const int mg      = wave & 3;
  const int ngbase  = (wave >> 2) * 2;

  for (int d = 0; d < NDIAG; ++d) {
    int jmin = d - (Hh - 1); if (jmin < 0) jmin = 0;
    int jmax = (d < Ww - 1) ? d : Ww - 1;
    const int j = jmin + cellslot;

    if (j <= jmax) {                         // this block has a cell on diagonal d
      const int i = d - j;
      float*       Hout = (d & 1) ? Hbuf1 : Hbuf0;
      const float* Hin  = (d & 1) ? Hbuf0 : Hbuf1;
      const float* htop  = (i > 0) ? (Hin + (size_t)j       * (B_ * HD)) : nullptr;
      const float* hleft = (j > 0) ? (Hin + (size_t)(j - 1) * (B_ * HD)) : nullptr;

      // Stage h_prev = 0.5*(h_top + h_left) as f16 (full 128x512 A tile).
      for (int idx = tid; idx < B_ * (HD / 4); idx += 256) {
        int bb = idx >> 7;
        int f  = (idx & 127) * 4;
        float4 t = htop  ? *(const float4*)(htop  + (size_t)bb * HD + f) : make_float4(0.f, 0.f, 0.f, 0.f);
        float4 l = hleft ? *(const float4*)(hleft + (size_t)bb * HD + f) : make_float4(0.f, 0.f, 0.f, 0.f);
        v4h_t hp;
        hp[0] = (_Float16)(0.5f * (t.x + l.x));
        hp[1] = (_Float16)(0.5f * (t.y + l.y));
        hp[2] = (_Float16)(0.5f * (t.z + l.z));
        hp[3] = (_Float16)(0.5f * (t.w + l.w));
        *(v4h_t*)(As + (size_t)bb * LDA + f) = hp;
      }
      __syncthreads();

      // Two A rows for the 2 M-frags of this wave.
      const _Float16* Arow0 = As + (size_t)(mg * 32      + l15) * LDA;
      const _Float16* Arow1 = As + (size_t)(mg * 32 + 16 + l15) * LDA;
      float* outcol = Hout + (size_t)j * (B_ * HD);

#pragma unroll
      for (int np = 0; np < 2; ++np) {
        const int ng    = ngbase + np;
        const int col0  = f0 + ng * 32;                  // slice-local 32-col group
        // B bases: gate g at Ut row (col + g*512); tile ni adds 16.
        const _Float16* Bb[3][2];
#pragma unroll
        for (int g = 0; g < 3; ++g)
#pragma unroll
          for (int ni = 0; ni < 2; ++ni)
            Bb[g][ni] = Ut + (size_t)(col0 + ni * 16 + l15 + g * HD) * HD;

        v8f acc[3][2][2];
#pragma unroll
        for (int g = 0; g < 3; ++g)
#pragma unroll
          for (int mi = 0; mi < 2; ++mi)
#pragma unroll
            for (int ni = 0; ni < 2; ++ni)
              acc[g][mi][ni] = (v8f){};

#pragma unroll 2
        for (int k = 0; k < 16; ++k) {
          const int k0 = k * 32 + khalf * 8;
          v16h a0 = frag_from(Arow0, k0);
          v16h a1 = frag_from(Arow1, k0);
          v16h bf[3][2];
#pragma unroll
          for (int g = 0; g < 3; ++g)
#pragma unroll
            for (int ni = 0; ni < 2; ++ni)
              bf[g][ni] = frag_from(Bb[g][ni], k0);
#pragma unroll
          for (int g = 0; g < 3; ++g)
#pragma unroll
            for (int ni = 0; ni < 2; ++ni) {
              acc[g][0][ni] = __builtin_amdgcn_wmma_f32_16x16x32_f16(false, a0, false, bf[g][ni], (short)0, acc[g][0][ni], false, false);
              acc[g][1][ni] = __builtin_amdgcn_wmma_f32_16x16x32_f16(false, a1, false, bf[g][ni], (short)0, acc[g][1][ni], false, false);
            }
        }

        // GRU elementwise update off the accumulators.
        // C/D layout: VGPR v holds M = v (lanes 0-15) / v+8 (lanes 16-31), N = lane&15.
#pragma unroll
        for (int ni = 0; ni < 2; ++ni) {
          const int fcol = col0 + ni * 16 + l15;
          const float vz = vfold[fcol],          cz = cfold[fcol];
          const float vr = vfold[fcol + HD],     cr = cfold[fcol + HD];
          const float vn = vfold[fcol + 2 * HD], cn = cfold[fcol + 2 * HD];
#pragma unroll
          for (int mi = 0; mi < 2; ++mi) {
#pragma unroll
            for (int vv = 0; vv < 8; ++vv) {
              const int brow = mg * 32 + mi * 16 + khalf * 8 + vv;
              const float xv = x[((size_t)brow * Hh + i) * Ww + j];
              const float ht = htop  ? htop [(size_t)brow * HD + fcol] : 0.f;
              const float hl = hleft ? hleft[(size_t)brow * HD + fcol] : 0.f;
              const float hp = 0.5f * (ht + hl);
              const float z  = 1.f / (1.f + __expf(-(xv * vz + cz + acc[0][mi][ni][vv])));
              const float r  = 1.f / (1.f + __expf(-(xv * vr + cr + acc[1][mi][ni][vv])));
              const float nn = tanhf(xv * vn + cn + r * acc[2][mi][ni][vv]);
              outcol[(size_t)brow * HD + fcol] = (1.f - z) * nn + z * hp;
            }
          }
        }
      }
    }

    // ---- device-wide barrier between diagonals ----
    __syncthreads();                                        // LDS + cell work done
    __builtin_amdgcn_fence(__ATOMIC_RELEASE, "agent");      // flush h writes (global_wb)
    if (tid == 0) {
      __hip_atomic_fetch_add(bar, 1u, __ATOMIC_RELAXED, __HIP_MEMORY_SCOPE_AGENT);
      const unsigned target = (unsigned)NBLK * (unsigned)(d + 1);
      while (__hip_atomic_load(bar, __ATOMIC_RELAXED, __HIP_MEMORY_SCOPE_AGENT) < target)
        __builtin_amdgcn_s_sleep(8);
    }
    __syncthreads();
    __builtin_amdgcn_fence(__ATOMIC_ACQUIRE, "agent");      // invalidate L0 per wave
  }
}

// ---- classifier head: (128,512) @ (512,10) + b ----------------------------
__global__ void head_kernel(const float* __restrict__ hfin, const float* __restrict__ W_out,
                            const float* __restrict__ b_out, float* __restrict__ out) {
  int t = blockIdx.x * blockDim.x + threadIdx.x;
  if (t >= B_ * NC) return;
  int b = t / NC, k = t % NC;
  float acc = b_out[k];
#pragma unroll 8
  for (int f = 0; f < HD; ++f) acc += hfin[(size_t)b * HD + f] * W_out[(size_t)f * NC + k];
  out[t] = acc;
}

// ---------------------------------------------------------------------------
extern "C" void kernel_launch(void* const* d_in, const int* in_sizes, int n_in,
                              void* d_out, int out_size, void* d_ws, size_t ws_size,
                              hipStream_t stream) {
  const float* x     = (const float*)d_in[0];
  const float* W_emb = (const float*)d_in[1];
  const float* b_emb = (const float*)d_in[2];
  const float* W_x   = (const float*)d_in[3];
  const float* b_x   = (const float*)d_in[4];
  const float* U_zr  = (const float*)d_in[5];
  const float* U_n   = (const float*)d_in[6];
  const float* W_out = (const float*)d_in[7];
  const float* b_out = (const float*)d_in[8];
  float* out = (float*)d_out;

  // Workspace layout (~16.3 MB total; 256B aligned).
  char* ws = (char*)d_ws;
  size_t off = 0;
  auto take = [&](size_t bytes) -> void* {
    void* p = ws + off;
    off = (off + bytes + 255) & ~(size_t)255;
    return p;
  };
  _Float16* Ut    = (_Float16*)take((size_t)THREE_HD * HD * sizeof(_Float16)); // 1.5 MB
  float*    vfold = (float*)take(THREE_HD * sizeof(float));
  float*    cfold = (float*)take(THREE_HD * sizeof(float));
  float*    Hbuf0 = (float*)take((size_t)Ww * B_ * HD * sizeof(float));        // 7.3 MB
  float*    Hbuf1 = (float*)take((size_t)Ww * B_ * HD * sizeof(float));        // 7.3 MB
  unsigned* bar   = (unsigned*)take(256);

  // Zero the device-wide barrier counter (graph-capture-legal async memset).
  hipMemsetAsync(bar, 0, sizeof(unsigned), stream);

  hipLaunchKernelGGL(fold_kernel, dim3((THREE_HD + 255) / 256), dim3(256), 0, stream,
                     W_emb, b_emb, W_x, b_x, vfold, cfold);
  hipLaunchKernelGGL(pack_u_kernel, dim3((THREE_HD * HD + 255) / 256), dim3(256), 0, stream,
                     U_zr, U_n, Ut);

  // One persistent launch runs all 55 anti-diagonals.
  hipLaunchKernelGGL(gru_wavefront_kernel, dim3(NBLK), dim3(256), 0, stream,
                     x, vfold, cfold, Ut, Hbuf0, Hbuf1, bar);

  // Final h is cell (27,27), written on diagonal 54 (even -> Hbuf0), column 27.
  const float* hfin = Hbuf0 + (size_t)(Ww - 1) * B_ * HD;
  hipLaunchKernelGGL(head_kernel, dim3((B_ * NC + 255) / 256), dim3(256), 0, stream,
                     hfin, W_out, b_out, out);
}
